// PolicyGNN_v3_40157944217754
// MI455X (gfx1250) — matrix operand order repaired and checked by reading, not compiled, not run on hardware
//
#include <hip/hip_runtime.h>
#include <hip/hip_bf16.h>
#include <stdint.h>

typedef __attribute__((ext_vector_type(16))) __bf16 v16bf;
typedef __attribute__((ext_vector_type(8)))  __bf16 v8bf;
typedef __attribute__((ext_vector_type(8)))  float  v8f;

#define HEADS 4
#define CCH   256   // channels per head

__device__ __forceinline__ float lrelu(float x, float s){ return x > 0.f ? x : s * x; }
// order-preserving float->uint key for atomicMax-based segment max
__device__ __forceinline__ unsigned f2key(float x){
  unsigned b = __float_as_uint(x);
  return (b & 0x80000000u) ? ~b : (b | 0x80000000u);
}
__device__ __forceinline__ float key2f(unsigned k){
  unsigned b = (k & 0x80000000u) ? (k & 0x7FFFFFFFu) : ~k;
  return __uint_as_float(b);
}

// ---------------------------------------------------------------- fp32 -> bf16
__global__ void k_cvt_bf16(const float* __restrict__ in, __bf16* __restrict__ out, long n){
  long i = (long)blockIdx.x * blockDim.x + threadIdx.x;
  long gs = (long)gridDim.x * blockDim.x;
  for (; i < n; i += gs) out[i] = (__bf16)in[i];
}

// fp32 [K][N] -> bf16 transposed [N][K]  (one-time per weight; lets the GEMM
// stage B tiles with pure b128 copies, no sub-word LDS transpose stores)
__global__ void k_cvt_bf16_t(const float* __restrict__ in, __bf16* __restrict__ out,
                             int K, int N){
  long n = (long)K * N;
  long i = (long)blockIdx.x * blockDim.x + threadIdx.x;
  long gs = (long)gridDim.x * blockDim.x;
  for (; i < n; i += gs) {
    int k = (int)(i / N);
    int c = (int)(i - (long)k * N);
    out[(size_t)c * K + k] = (__bf16)in[i];
  }
}

// ---------------------------------------------------------------- WMMA GEMM
// C[M,N](f32) = A[M,K](bf16, row-major) * Bt[N,K](bf16, pre-transposed) + bias.
// Block tile 64x128, BK=32, 256 threads = 8 waves in 2(M) x 4(N) grid.
// Each wave owns a 32x32 slab: 2 A-frags x 2 B-frags -> 4 WMMAs per k-step,
// 8 ds_load_b128 per 4 v_wmma_f32_16x16x32_bf16.
#define BM 64
#define BN 128
#define BK 32
#define LDP (BK + 8)   // 40 elems = 80B row stride, keeps 16B alignment

// A fragment (16x32): lane l<16 -> row m=l, K=0..15;
// lane 16+l -> same row set, K=16..31
__device__ __forceinline__ v16bf fragA(const __bf16* rowp, int kA){
  v8bf lo = *reinterpret_cast<const v8bf*>(rowp + kA);
  v8bf hi = *reinterpret_cast<const v8bf*>(rowp + kA + 8);
  return __builtin_shufflevector(lo, hi, 0,1,2,3,4,5,6,7,8,9,10,11,12,13,14,15);
}
// B fragment (32x16): lane l<16 -> col n=l, K=0..15; lane 16+l -> K=16..31
__device__ __forceinline__ v16bf fragB(const __bf16* rowp, int kB){
  v8bf lo = *reinterpret_cast<const v8bf*>(rowp + kB);
  v8bf hi = *reinterpret_cast<const v8bf*>(rowp + kB + 8);
  return __builtin_shufflevector(lo, hi, 0,1,2,3,4,5,6,7,8,9,10,11,12,13,14,15);
}

__global__ __launch_bounds__(256)
void k_gemm_bf16_wmma(const __bf16* __restrict__ A, const __bf16* __restrict__ Bt,
                      const float* __restrict__ bias, float* __restrict__ C,
                      int M, int N, int K, int act){
  __shared__ __align__(16) __bf16 sA[BM][LDP];   // [m][k]
  __shared__ __align__(16) __bf16 sB[BN][LDP];   // [n][k]
  const int tid  = threadIdx.x;
  const int m0   = blockIdx.y * BM;
  const int n0   = blockIdx.x * BN;
  const int w    = tid >> 5;
  const int lane = tid & 31;
  const int mBase = (w >> 2) * 32;   // 2 waves along M
  const int nBase = (w & 3) * 32;    // 4 waves along N

  v8f c00 = {}, c01 = {}, c10 = {}, c11 = {};

  // staging maps: A 64x32 -> 8 bf16/thread; B 128x32 -> 2 x 8 bf16/thread
  const int ar = tid >> 2;           // 0..63
  const int ac = (tid & 3) * 8;      // 0..24

  for (int k0 = 0; k0 < K; k0 += BK) {
    {
      const int grow = m0 + ar;
      v8bf av = {};
      if (grow < M)
        av = *reinterpret_cast<const v8bf*>(A + (size_t)grow * K + k0 + ac);
      *reinterpret_cast<v8bf*>(&sA[ar][ac]) = av;
    }
    #pragma unroll
    for (int i = 0; i < 2; ++i) {
      const int ch = tid + i * 256;
      const int rn = ch >> 2;            // 0..127
      const int ck = (ch & 3) * 8;       // 0..24
      *reinterpret_cast<v8bf*>(&sB[rn][ck]) =
        *reinterpret_cast<const v8bf*>(Bt + (size_t)(n0 + rn) * K + k0 + ck);
    }
    __syncthreads();

    const int half = lane >> 4;
    const int kA = half * 16;
    const int kB = half * 16;
    const int mr = mBase + (lane & 15);
    const int nc = nBase + (lane & 15);

    v16bf a0 = fragA(&sA[mr][0],      kA);
    v16bf a1 = fragA(&sA[mr + 16][0], kA);
    v16bf b0 = fragB(&sB[nc][0],      kB);
    v16bf b1 = fragB(&sB[nc + 16][0], kB);

    c00 = __builtin_amdgcn_wmma_f32_16x16x32_bf16(false, a0, false, b0, (short)0, c00, false, false);
    c01 = __builtin_amdgcn_wmma_f32_16x16x32_bf16(false, a0, false, b1, (short)0, c01, false, false);
    c10 = __builtin_amdgcn_wmma_f32_16x16x32_bf16(false, a1, false, b0, (short)0, c10, false, false);
    c11 = __builtin_amdgcn_wmma_f32_16x16x32_bf16(false, a1, false, b1, (short)0, c11, false, false);
    __syncthreads();
  }

  // C layout: lane l holds col n=(l&15); VGPR r -> row r + 8*(l>>4)
  const int col0 = n0 + nBase + (lane & 15);
  const int col1 = col0 + 16;
  const int rb   = m0 + mBase + ((lane >> 4) * 8);
  const float bi0 = bias[col0];
  const float bi1 = bias[col1];
  #pragma unroll
  for (int r = 0; r < 8; ++r) {
    const int row = rb + r;
    if (row < M) {
      float v0 = c00[r] + bi0;
      float v1 = c01[r] + bi1;
      if (act == 1) { v0 = lrelu(v0, 0.01f); v1 = lrelu(v1, 0.01f); }
      C[(size_t)row * N + col0] = v0;
      C[(size_t)row * N + col1] = v1;
    }
    const int row2 = row + 16;
    if (row2 < M) {
      float v0 = c10[r] + bi0;
      float v1 = c11[r] + bi1;
      if (act == 1) { v0 = lrelu(v0, 0.01f); v1 = lrelu(v1, 0.01f); }
      C[(size_t)row2 * N + col0] = v0;
      C[(size_t)row2 * N + col1] = v1;
    }
  }
}

// ---------------------------------------------------------------- edge phase
// one wave per (edge, head): score = att . leaky_relu(xl[s]+xr[d], 0.2), seg-max
__global__ __launch_bounds__(256)
void k_edge_scores(const int* __restrict__ src, const int* __restrict__ dst,
                   const float* __restrict__ xl, const float* __restrict__ xr,
                   const float* __restrict__ att,
                   float* __restrict__ sc, unsigned* __restrict__ nmax,
                   int Eraw, int T, int H){
  const int w = blockIdx.x * 8 + (threadIdx.x >> 5);
  const int lane = threadIdx.x & 31;
  if (w >= T * H) return;
  const int e = w / H, hh = w - e * H;
  int s, d;
  if (e < Eraw) { s = src[e]; d = dst[e]; } else { s = e - Eraw; d = s; }
  const int stride = H * CCH;
  const float* pl = xl + (long)s * stride + hh * CCH;
  const float* pr = xr + (long)d * stride + hh * CCH;
  const float* pa = att + hh * CCH;
  float acc = 0.f;
  #pragma unroll
  for (int i = 0; i < CCH / 32; ++i) {
    const int c = lane + i * 32;
    acc += pa[c] * lrelu(pl[c] + pr[c], 0.2f);
  }
  #pragma unroll
  for (int off = 16; off > 0; off >>= 1) acc += __shfl_xor(acc, off, 32);
  if (lane == 0) {
    sc[w] = acc;
    atomicMax(&nmax[d * H + hh], f2key(acc));
  }
}

__global__ void k_edge_exp(const int* __restrict__ dst, float* __restrict__ sc,
                           const unsigned* __restrict__ nmax, float* __restrict__ nden,
                           int Eraw, int T, int H){
  const int idx = blockIdx.x * blockDim.x + threadIdx.x;
  if (idx >= T * H) return;
  const int e = idx / H, hh = idx - e * H;
  const int d = (e < Eraw) ? dst[e] : (e - Eraw);
  const float m = key2f(nmax[d * H + hh]);
  const float ex = __expf(sc[idx] - m);
  sc[idx] = ex;
  __hip_atomic_fetch_add(&nden[d * H + hh], ex, __ATOMIC_RELAXED, __HIP_MEMORY_SCOPE_AGENT);
}

__global__ __launch_bounds__(256)
void k_edge_agg(const int* __restrict__ src, const int* __restrict__ dst,
                const float* __restrict__ xl, const float* __restrict__ sc,
                const float* __restrict__ nden, float* __restrict__ agg,
                int Eraw, int T, int H){
  const int w = blockIdx.x * 8 + (threadIdx.x >> 5);
  const int lane = threadIdx.x & 31;
  if (w >= T * H) return;
  const int e = w / H, hh = w - e * H;
  int s, d;
  if (e < Eraw) { s = src[e]; d = dst[e]; } else { s = e - Eraw; d = s; }
  const float alpha = sc[w] / nden[d * H + hh];
  const int stride = H * CCH;
  const float* pl = xl + (long)s * stride + hh * CCH;
  float* po = agg + (long)d * stride + hh * CCH;
  #pragma unroll
  for (int i = 0; i < CCH / 32; ++i) {
    const int c = lane + i * 32;
    __hip_atomic_fetch_add(&po[c], alpha * pl[c], __ATOMIC_RELAXED, __HIP_MEMORY_SCOPE_AGENT);
  }
}

// ---------------------------------------------------------------- epilogues
__global__ void k_elu_bias_bf16(const float* __restrict__ a, const float* __restrict__ bias,
                                __bf16* __restrict__ o, long n, int stride){
  long i = (long)blockIdx.x * blockDim.x + threadIdx.x;
  long gs = (long)gridDim.x * blockDim.x;
  for (; i < n; i += gs) {
    float v = a[i] + bias[i % stride];
    v = v > 0.f ? v : (__expf(v) - 1.f);   // elu
    o[i] = (__bf16)v;
  }
}

__global__ void k_bias_f32(const float* __restrict__ a, const float* __restrict__ bias,
                           float* __restrict__ o, long n, int stride){
  long i = (long)blockIdx.x * blockDim.x + threadIdx.x;
  long gs = (long)gridDim.x * blockDim.x;
  for (; i < n; i += gs) o[i] = a[i] + bias[i % stride];
}

// logits = (leaky(h3) + h) . W_out + b_out ; one wave per node
__global__ __launch_bounds__(256)
void k_final(const float* __restrict__ h3, const float* __restrict__ h,
             const float* __restrict__ Wout, const float* __restrict__ bout,
             float* __restrict__ out, int Nn){
  const int w = blockIdx.x * 8 + (threadIdx.x >> 5);
  const int lane = threadIdx.x & 31;
  if (w >= Nn) return;
  const float* p3 = h3 + (long)w * CCH;
  const float* ph = h  + (long)w * CCH;
  float acc = 0.f;
  #pragma unroll
  for (int i = 0; i < CCH / 32; ++i) {
    const int c = lane + i * 32;
    acc += (lrelu(p3[c], 0.01f) + ph[c]) * Wout[c];
  }
  #pragma unroll
  for (int off = 16; off > 0; off >>= 1) acc += __shfl_xor(acc, off, 32);
  if (lane == 0) out[w] = acc + bout[0];
}

// ---------------------------------------------------------------- launcher
extern "C" void kernel_launch(void* const* d_in, const int* in_sizes, int n_in,
                              void* d_out, int out_size, void* d_ws, size_t ws_size,
                              hipStream_t stream) {
  const float* x     = (const float*)d_in[0];
  const int*   src   = (const int*)  d_in[1];
  const int*   dst   = (const int*)  d_in[2];
  const float* W_in  = (const float*)d_in[3];  const float* b_in  = (const float*)d_in[4];
  const float* Wl1   = (const float*)d_in[5];  const float* bl1   = (const float*)d_in[6];
  const float* Wr1   = (const float*)d_in[7];  const float* br1   = (const float*)d_in[8];
  const float* att1  = (const float*)d_in[9];  const float* bias1 = (const float*)d_in[10];
  const float* Wl2   = (const float*)d_in[11]; const float* bl2   = (const float*)d_in[12];
  const float* Wr2   = (const float*)d_in[13]; const float* br2   = (const float*)d_in[14];
  const float* att2  = (const float*)d_in[15]; const float* bias2 = (const float*)d_in[16];
  const float* Wl3   = (const float*)d_in[17]; const float* bl3   = (const float*)d_in[18];
  const float* Wr3   = (const float*)d_in[19]; const float* br3   = (const float*)d_in[20];
  const float* att3  = (const float*)d_in[21]; const float* bias3 = (const float*)d_in[22];
  const float* W_out = (const float*)d_in[23]; const float* b_out = (const float*)d_in[24];
  float* out = (float*)d_out;

  const int DIN  = 64;
  const int Nn   = in_sizes[0] / DIN;
  const int Eraw = in_sizes[1];
  const int T    = Eraw + Nn;            // with self loops
  const int H    = HEADS;
  const int F    = H * CCH;              // 1024

  // workspace carve
  char* base = (char*)d_ws; size_t off = 0;
  auto alloc = [&](size_t bytes) -> void* {
    void* q = base + off; off += (bytes + 255) & ~(size_t)255; return q;
  };
  float*    h    = (float*)   alloc((size_t)Nn * CCH * 4);
  float*    h3   = (float*)   alloc((size_t)Nn * CCH * 4);
  float*    xl   = (float*)   alloc((size_t)Nn * F * 4);
  float*    xr   = (float*)   alloc((size_t)Nn * F * 4);
  float*    agg  = (float*)   alloc((size_t)Nn * F * 4);
  __bf16*   hb   = (__bf16*)  alloc((size_t)Nn * F * 2);
  __bf16*   wbl  = (__bf16*)  alloc((size_t)F * F * 2);   // transposed weights
  __bf16*   wbr  = (__bf16*)  alloc((size_t)F * F * 2);
  float*    sc   = (float*)   alloc((size_t)T * H * 4);
  unsigned* nmax = (unsigned*)alloc((size_t)Nn * H * 4);
  float*    nden = (float*)   alloc((size_t)Nn * H * 4);
  (void)ws_size; (void)n_in; (void)out_size;

  dim3 blk(256);
  auto cvt = [&](const float* in, __bf16* o, long n) {
    int g = (int)((n + 255) / 256); if (g > 4096) g = 4096;
    k_cvt_bf16<<<dim3(g), blk, 0, stream>>>(in, o, n);
  };
  auto cvtT = [&](const float* in, __bf16* o, int K, int N) {
    long n = (long)K * N;
    int g = (int)((n + 255) / 256); if (g > 4096) g = 4096;
    k_cvt_bf16_t<<<dim3(g), blk, 0, stream>>>(in, o, K, N);
  };
  auto edge_phase = [&](int Hc, const float* att, const float* xl_, const float* xr_,
                        float* agg_) {
    hipMemsetAsync(agg_, 0, (size_t)Nn * Hc * CCH * 4, stream);
    hipMemsetAsync(nmax, 0, (size_t)Nn * Hc * 4, stream);
    hipMemsetAsync(nden, 0, (size_t)Nn * Hc * 4, stream);
    const long tasks = (long)T * Hc;
    const int gw = (int)((tasks + 7) / 8);
    const int gt = (int)((tasks + 255) / 256);
    k_edge_scores<<<dim3(gw), blk, 0, stream>>>(src, dst, xl_, xr_, att, sc, nmax, Eraw, T, Hc);
    k_edge_exp   <<<dim3(gt), blk, 0, stream>>>(dst, sc, nmax, nden, Eraw, T, Hc);
    k_edge_agg   <<<dim3(gw), blk, 0, stream>>>(src, dst, xl_, sc, nden, agg_, Eraw, T, Hc);
  };

  const dim3 gIn (CCH / BN, (Nn + BM - 1) / BM);   // N=256  -> 2 x 313
  const dim3 gBig(F   / BN, (Nn + BM - 1) / BM);   // N=1024 -> 8 x 313

  // ---- input layer: h = leaky(x @ W_in + b_in)
  cvt(x, hb, (long)Nn * DIN);
  cvtT(W_in, wbl, DIN, CCH);
  k_gemm_bf16_wmma<<<gIn, blk, 0, stream>>>(hb, wbl, b_in, h, Nn, CCH, DIN, 1);
  cvt(h, hb, (long)Nn * CCH);

  // ---- conv1 (in 256 -> 4x256 concat)
  cvtT(Wl1, wbl, CCH, F);
  cvtT(Wr1, wbr, CCH, F);
  k_gemm_bf16_wmma<<<gBig, blk, 0, stream>>>(hb, wbl, bl1, xl, Nn, F, CCH, 0);
  k_gemm_bf16_wmma<<<gBig, blk, 0, stream>>>(hb, wbr, br1, xr, Nn, F, CCH, 0);
  edge_phase(H, att1, xl, xr, agg);
  { long n = (long)Nn * F; int g = (int)((n + 255) / 256); if (g > 4096) g = 4096;
    k_elu_bias_bf16<<<dim3(g), blk, 0, stream>>>(agg, bias1, hb, n, F); }

  // ---- conv2 (in 1024 -> 4x256 concat)
  cvtT(Wl2, wbl, F, F);
  cvtT(Wr2, wbr, F, F);
  k_gemm_bf16_wmma<<<gBig, blk, 0, stream>>>(hb, wbl, bl2, xl, Nn, F, F, 0);
  k_gemm_bf16_wmma<<<gBig, blk, 0, stream>>>(hb, wbr, br2, xr, Nn, F, F, 0);
  edge_phase(H, att2, xl, xr, agg);
  { long n = (long)Nn * F; int g = (int)((n + 255) / 256); if (g > 4096) g = 4096;
    k_elu_bias_bf16<<<dim3(g), blk, 0, stream>>>(agg, bias2, hb, n, F); }

  // ---- conv3 (in 1024 -> 1x256, no concat)
  cvtT(Wl3, wbl, F, CCH);
  cvtT(Wr3, wbr, F, CCH);
  k_gemm_bf16_wmma<<<gIn, blk, 0, stream>>>(hb, wbl, bl3, xl, Nn, CCH, F, 0);
  k_gemm_bf16_wmma<<<gIn, blk, 0, stream>>>(hb, wbr, br3, xr, Nn, CCH, F, 0);
  edge_phase(1, att3, xl, xr, agg);
  { long n = (long)Nn * CCH; int g = (int)((n + 255) / 256); if (g > 4096) g = 4096;
    k_bias_f32<<<dim3(g), blk, 0, stream>>>(agg, bias3, h3, n, CCH); }

  // ---- output: logits = (leaky(h3) + h) @ W_out + b_out
  k_final<<<dim3((Nn + 7) / 8), blk, 0, stream>>>(h3, h, W_out, b_out, out, Nn);
}
